// PerceiverAttention_35708358099620
// MI455X (gfx1250) — matrix operand order repaired and checked by reading, not compiled, not run on hardware
//
#include <hip/hip_runtime.h>
#include <hip/hip_bf16.h>
#include <math.h>

// ---------------- problem sizes ----------------
#define B_   4
#define SX_  4096
#define SL_  256
#define D_   1024
#define H_   16
#define DH_  64
#define SKV_ (SX_ + SL_)   // 4352

typedef __attribute__((ext_vector_type(16))) _Float16 v16h;
typedef __attribute__((ext_vector_type(8)))  _Float16 v8h;
typedef __attribute__((ext_vector_type(8)))  float    v8f;

union V16 { v16h v; struct { v8h lo, hi; } h; };

__device__ __forceinline__ v8f wmma_f16(v16h a, v16h b, v8f c) {
  // D = A(16x32 f16) x B(32x16 f16) + C(16x16 f32)
  return __builtin_amdgcn_wmma_f32_16x16x32_f16(
      /*neg_a=*/false, a, /*neg_b=*/false, b,
      /*c_mod=*/(short)0, c, /*reuse_a=*/false, /*reuse_b=*/false);
}

// A fragment (16x32, row-major source). row_ptr points at &A[row][kk], row = lane&15.
// ISA A-layout: lanes 0-15 hold K {0..7,16..23}, lanes 16-31 hold K {8..15,24..31}.
__device__ __forceinline__ v16h load_a(const _Float16* row_ptr, int lane) {
  int k0 = (lane & 16) ? 8 : 0;
  V16 u;
  u.h.lo = *(const v8h*)(row_ptr + k0);
  u.h.hi = *(const v8h*)(row_ptr + 16 + k0);
  return u.v;
}

// B fragment (32x16) loaded from B^T stored row-major [N][K]. row_ptr = &Bt[n][kk],
// n = lane&15. B-layout (matches A): lanes 0-15 hold K {0..7,16..23}, lanes 16-31 K {8..15,24..31}.
__device__ __forceinline__ v16h load_b(const _Float16* row_ptr, int lane) {
  int k0 = (lane & 16) ? 8 : 0;
  V16 u;
  u.h.lo = *(const v8h*)(row_ptr + k0);
  u.h.hi = *(const v8h*)(row_ptr + 16 + k0);
  return u.v;
}

__device__ __forceinline__ float wave16_sum(float v) {
  v += __shfl_xor(v, 1, 32);
  v += __shfl_xor(v, 2, 32);
  v += __shfl_xor(v, 4, 32);
  v += __shfl_xor(v, 8, 32);
  return v;
}
__device__ __forceinline__ float wave16_max(float v) {
  v = fmaxf(v, __shfl_xor(v, 1, 32));
  v = fmaxf(v, __shfl_xor(v, 2, 32));
  v = fmaxf(v, __shfl_xor(v, 4, 32));
  v = fmaxf(v, __shfl_xor(v, 8, 32));
  return v;
}
__device__ __forceinline__ float wave32_sum(float v) {
  v += __shfl_xor(v, 16, 32);
  v += __shfl_xor(v, 8, 32);
  v += __shfl_xor(v, 4, 32);
  v += __shfl_xor(v, 2, 32);
  v += __shfl_xor(v, 1, 32);
  return v;
}

// ------------- weight convert + transpose: Wt[n][k] = (f16)W[k][n] -------------
__global__ void k_transpose_f16(const float* __restrict__ W, _Float16* __restrict__ Wt,
                                int K, int N) {
  int n = blockIdx.x;
  for (int k = threadIdx.x; k < K; k += blockDim.x)
    Wt[(size_t)n * K + k] = (_Float16)W[(size_t)k * N + n];
}

// ------------- layernorm over D=1024, one row per block (256 thr) -------------
__global__ void k_layernorm_f16(const float* __restrict__ X, const float* __restrict__ g,
                                const float* __restrict__ bta, _Float16* __restrict__ out) {
  __shared__ float red[8];
  int row = blockIdx.x;
  const float* x = X + (size_t)row * D_;
  float v[4];
  float s = 0.f;
#pragma unroll
  for (int i = 0; i < 4; i++) { v[i] = x[threadIdx.x + 256 * i]; s += v[i]; }
  s = wave32_sum(s);
  int wid = threadIdx.x >> 5;
  if ((threadIdx.x & 31) == 0) red[wid] = s;
  __syncthreads();
  float tot = 0.f;
#pragma unroll
  for (int w = 0; w < 8; w++) tot += red[w];
  float mu = tot * (1.0f / D_);
  __syncthreads();
  float s2 = 0.f;
#pragma unroll
  for (int i = 0; i < 4; i++) { float d = v[i] - mu; s2 += d * d; }
  s2 = wave32_sum(s2);
  if ((threadIdx.x & 31) == 0) red[wid] = s2;
  __syncthreads();
  float t2 = 0.f;
#pragma unroll
  for (int w = 0; w < 8; w++) t2 += red[w];
  float rs = rsqrtf(t2 * (1.0f / D_) + 1e-5f);
#pragma unroll
  for (int i = 0; i < 4; i++) {
    int c = threadIdx.x + 256 * i;
    out[(size_t)row * D_ + c] = (_Float16)((v[i] - mu) * rs * g[c] + bta[c]);
  }
}

// ------------- kv GEMM: [B*Skv,1024] x Wkv_t[2048,1024]^T.
// Each wave owns a 16x64 tile (A-frag reused across 4 WMMAs); the 64-col group
// is exactly one k head or one v head, so the k RMS-norm is wave-local (no LDS).
// grid (B*Skv/64, 32), block 128 (4 waves row-split). -------------
__global__ void k_gemm_kv(const _Float16* __restrict__ xn, const _Float16* __restrict__ lnl,
                          const _Float16* __restrict__ Wkv_t, const float* __restrict__ gamma,
                          _Float16* __restrict__ kh, _Float16* __restrict__ vt) {
  int lane = threadIdx.x & 31;
  int wv = threadIdx.x >> 5;                 // 0..3 : row-tile within block
  int r0 = blockIdx.x * 64 + wv * 16;        // row tile within B*Skv
  int ng = blockIdx.y;                       // 0..31: 64-col group (0..15 -> k heads)

  // per-lane A row source (xn vs ln)
  int gr = r0 + (lane & 15);
  int b = gr / SKV_;
  int rr = gr - b * SKV_;
  const _Float16* arow = (rr < SX_)
      ? (xn + ((size_t)b * SX_ + rr) * D_)
      : (lnl + ((size_t)b * SL_ + (rr - SX_)) * D_);
  const _Float16* br0 = Wkv_t + (size_t)(ng * 64 +  0 + (lane & 15)) * D_;
  const _Float16* br1 = Wkv_t + (size_t)(ng * 64 + 16 + (lane & 15)) * D_;
  const _Float16* br2 = Wkv_t + (size_t)(ng * 64 + 32 + (lane & 15)) * D_;
  const _Float16* br3 = Wkv_t + (size_t)(ng * 64 + 48 + (lane & 15)) * D_;

  v8f acc[4] = {};
  for (int kk = 0; kk < D_; kk += 32) {
    v16h af = load_a(arow + kk, lane);
    acc[0] = wmma_f16(af, load_b(br0 + kk, lane), acc[0]);
    acc[1] = wmma_f16(af, load_b(br1 + kk, lane), acc[1]);
    acc[2] = wmma_f16(af, load_b(br2 + kk, lane), acc[2]);
    acc[3] = wmma_f16(af, load_b(br3 + kk, lane), acc[3]);
  }

  int rbase = (lane & 16) ? 8 : 0;           // this lane holds rows rbase..rbase+7
  int gr0 = r0 + rbase;
  int bb = gr0 / SKV_;
  int kr0 = gr0 - bb * SKV_;
  if (ng < H_) {
    // k path: per-row RMS norm over the head's 64 cols (wave-local)
    float sc[8];
#pragma unroll
    for (int i = 0; i < 8; i++) {
      float t = acc[0][i] * acc[0][i] + acc[1][i] * acc[1][i] +
                acc[2][i] * acc[2][i] + acc[3][i] * acc[3][i];
      float tot = wave16_sum(t);
      float nrm = sqrtf(tot * (1.0f / DH_));
      sc[i] = 1.0f / fmaxf(nrm, 1e-8f);
    }
#pragma unroll
    for (int j = 0; j < 4; j++) {
      int dh = j * 16 + (lane & 15);
      float gv = gamma[dh];
      _Float16* kout = kh + (((size_t)bb * H_ + ng) * SKV_ + kr0) * DH_ + dh;
#pragma unroll
      for (int i = 0; i < 8; i++)
        kout[(size_t)i * DH_] = (_Float16)(acc[j][i] * sc[i] * gv);
    }
  } else {
    // v path: write transposed v_t[b][h][dh][skv], 16B per lane per subtile
    int h = ng - H_;
#pragma unroll
    for (int j = 0; j < 4; j++) {
      int dh = j * 16 + (lane & 15);
      v8h pk;
#pragma unroll
      for (int i = 0; i < 8; i++) pk[i] = (_Float16)acc[j][i];
      *(v8h*)(vt + (((size_t)bb * H_ + h) * DH_ + dh) * SKV_ + kr0) = pk;
    }
  }
}

// ------------- q GEMM: [B*SL,1024] x Wq_t[1024,1024]^T, fused wave-local RMSnorm.
//               grid (B*SL/64, 16), block 128 -------------
__global__ void k_gemm_q(const _Float16* __restrict__ lnl, const _Float16* __restrict__ Wq_t,
                         const float* __restrict__ gamma, _Float16* __restrict__ qh) {
  int lane = threadIdx.x & 31;
  int wv = threadIdx.x >> 5;
  int r0 = blockIdx.x * 64 + wv * 16;
  int h = blockIdx.y;                        // head == 64-col group

  const _Float16* arow = lnl + (size_t)(r0 + (lane & 15)) * D_;
  const _Float16* br0 = Wq_t + (size_t)(h * 64 +  0 + (lane & 15)) * D_;
  const _Float16* br1 = Wq_t + (size_t)(h * 64 + 16 + (lane & 15)) * D_;
  const _Float16* br2 = Wq_t + (size_t)(h * 64 + 32 + (lane & 15)) * D_;
  const _Float16* br3 = Wq_t + (size_t)(h * 64 + 48 + (lane & 15)) * D_;

  v8f acc[4] = {};
  for (int kk = 0; kk < D_; kk += 32) {
    v16h af = load_a(arow + kk, lane);
    acc[0] = wmma_f16(af, load_b(br0 + kk, lane), acc[0]);
    acc[1] = wmma_f16(af, load_b(br1 + kk, lane), acc[1]);
    acc[2] = wmma_f16(af, load_b(br2 + kk, lane), acc[2]);
    acc[3] = wmma_f16(af, load_b(br3 + kk, lane), acc[3]);
  }

  int rbase = (lane & 16) ? 8 : 0;
  int gr0 = r0 + rbase;
  int bb = gr0 / SL_;
  int sl0 = gr0 - bb * SL_;
  float sc[8];
#pragma unroll
  for (int i = 0; i < 8; i++) {
    float t = acc[0][i] * acc[0][i] + acc[1][i] * acc[1][i] +
              acc[2][i] * acc[2][i] + acc[3][i] * acc[3][i];
    float tot = wave16_sum(t);
    float nrm = sqrtf(tot * (1.0f / DH_));
    sc[i] = 1.0f / fmaxf(nrm, 1e-8f);
  }
#pragma unroll
  for (int j = 0; j < 4; j++) {
    int dh = j * 16 + (lane & 15);
    float gv = gamma[dh];
    _Float16* qout = qh + (((size_t)bb * H_ + h) * SL_ + sl0) * DH_ + dh;
#pragma unroll
    for (int i = 0; i < 8; i++)
      qout[(size_t)i * DH_] = (_Float16)(acc[j][i] * sc[i] * gv);
  }
}

// ------------- flash attention: one (b,h,16-query-tile) per block, 4 waves.
//               chunks of 64 kv rows, online softmax. -------------
__global__ void k_attn(const _Float16* __restrict__ qh, const _Float16* __restrict__ kh,
                       const _Float16* __restrict__ vt, const float* __restrict__ mask_x,
                       const float* __restrict__ mask_latent, _Float16* __restrict__ yh) {
  __shared__ float redm[4][16];
  __shared__ float reds[4][16];
  __shared__ _Float16 P[16][64];
  int lane = threadIdx.x & 31;
  int wv = threadIdx.x >> 5;
  int qt = blockIdx.x & 15;
  int h = (blockIdx.x >> 4) & 15;
  int b = blockIdx.x >> 8;

  const _Float16* qbase = qh + (((size_t)b * H_ + h) * SL_ + qt * 16) * DH_;
  const _Float16* kbase = kh + ((size_t)b * H_ + h) * (size_t)SKV_ * DH_;
  const _Float16* vbase = vt + ((size_t)b * H_ + h) * (size_t)DH_ * SKV_;

  int rbase = (lane & 16) ? 8 : 0;
  // Q fragments for this tile (reused across all chunks)
  const _Float16* qrow = qbase + (size_t)(lane & 15) * DH_;
  v16h qf0 = load_a(qrow, lane);
  v16h qf1 = load_a(qrow + 32, lane);

  float qm[8];
#pragma unroll
  for (int i = 0; i < 8; i++) qm[i] = mask_latent[b * SL_ + qt * 16 + rbase + i];

  float m8[8], l8[8];
#pragma unroll
  for (int i = 0; i < 8; i++) { m8[i] = -3.0e38f; l8[i] = 0.f; }
  v8f oacc = {};

  int dh = wv * 16 + (lane & 15);
  const _Float16* vrow = vbase + (size_t)dh * SKV_;

  for (int c0 = 0; c0 < SKV_; c0 += 64) {
    // ---- S = Q K^T for this wave's 16 kv columns ----
    int kvr = c0 + wv * 16 + (lane & 15);
    const _Float16* krow = kbase + (size_t)kvr * DH_;
    v8f s = {};
    s = wmma_f16(qf0, load_b(krow, lane), s);
    s = wmma_f16(qf1, load_b(krow + 32, lane), s);

    float kvm = (kvr < SX_) ? mask_x[b * SX_ + kvr]
                            : mask_latent[b * SL_ + (kvr - SX_)];
    float sv[8];
#pragma unroll
    for (int i = 0; i < 8; i++)
      sv[i] = s[i] * 0.125f + (1.0f - kvm * qm[i]) * (-3.0e38f);

    // ---- chunk row max across 64 cols (4 waves) ----
    float rm[8];
#pragma unroll
    for (int i = 0; i < 8; i++) rm[i] = wave16_max(sv[i]);
    if (lane == 0)  { for (int i = 0; i < 8; i++) redm[wv][i] = rm[i]; }
    if (lane == 16) { for (int i = 0; i < 8; i++) redm[wv][8 + i] = rm[i]; }
    __syncthreads();

    float mnew[8], pv[8], psum[8];
#pragma unroll
    for (int i = 0; i < 8; i++) {
      int row = rbase + i;
      float cm = fmaxf(fmaxf(redm[0][row], redm[1][row]), fmaxf(redm[2][row], redm[3][row]));
      mnew[i] = fmaxf(m8[i], cm);
      pv[i] = __expf(sv[i] - mnew[i]);
      psum[i] = wave16_sum(pv[i]);
    }
    if (lane == 0)  { for (int i = 0; i < 8; i++) reds[wv][i] = psum[i]; }
    if (lane == 16) { for (int i = 0; i < 8; i++) reds[wv][8 + i] = psum[i]; }
    __syncthreads();

#pragma unroll
    for (int i = 0; i < 8; i++) {
      int row = rbase + i;
      float csum = reds[0][row] + reds[1][row] + reds[2][row] + reds[3][row];
      float alpha = __expf(m8[i] - mnew[i]);
      l8[i] = l8[i] * alpha + csum;
      oacc[i] *= alpha;
      m8[i] = mnew[i];
      P[row][wv * 16 + (lane & 15)] = (_Float16)pv[i];
    }
    __syncthreads();

    // ---- O += P V  (A frags from LDS, B frags from transposed v) ----
    const _Float16* prow = &P[lane & 15][0];
    v16h pf0 = load_a(prow, lane);
    v16h pf1 = load_a(prow + 32, lane);
    v16h vf0 = load_b(vrow + c0, lane);
    v16h vf1 = load_b(vrow + c0 + 32, lane);
    oacc = wmma_f16(pf0, vf0, oacc);
    oacc = wmma_f16(pf1, vf1, oacc);
    __syncthreads();
  }

  // epilogue: divide by softmax denom, write y[b][sl][h*64+dh] as f16
#pragma unroll
  for (int i = 0; i < 8; i++) {
    int row = rbase + i;
    float val = oacc[i] / l8[i];
    yh[((size_t)(b * SL_ + qt * 16 + row)) * D_ + h * DH_ + dh] = (_Float16)val;
  }
}

// ------------- out projection: y_h[1024,1024] x Wo_t^T -> f32 d_out.
//               16x64 per wave, grid (B*SL/64, D/64), block 128 -------------
__global__ void k_gemm_out(const _Float16* __restrict__ yh, const _Float16* __restrict__ Wo_t,
                           float* __restrict__ out) {
  int lane = threadIdx.x & 31;
  int wv = threadIdx.x >> 5;
  int r0 = blockIdx.x * 64 + wv * 16;
  int c0 = blockIdx.y * 64;
  const _Float16* arow = yh + (size_t)(r0 + (lane & 15)) * D_;
  const _Float16* br0 = Wo_t + (size_t)(c0 +  0 + (lane & 15)) * D_;
  const _Float16* br1 = Wo_t + (size_t)(c0 + 16 + (lane & 15)) * D_;
  const _Float16* br2 = Wo_t + (size_t)(c0 + 32 + (lane & 15)) * D_;
  const _Float16* br3 = Wo_t + (size_t)(c0 + 48 + (lane & 15)) * D_;

  v8f acc[4] = {};
  for (int kk = 0; kk < D_; kk += 32) {
    v16h af = load_a(arow + kk, lane);
    acc[0] = wmma_f16(af, load_b(br0 + kk, lane), acc[0]);
    acc[1] = wmma_f16(af, load_b(br1 + kk, lane), acc[1]);
    acc[2] = wmma_f16(af, load_b(br2 + kk, lane), acc[2]);
    acc[3] = wmma_f16(af, load_b(br3 + kk, lane), acc[3]);
  }
  int rbase = (lane & 16) ? 8 : 0;
#pragma unroll
  for (int j = 0; j < 4; j++) {
    int col = c0 + j * 16 + (lane & 15);
#pragma unroll
    for (int i = 0; i < 8; i++)
      out[(size_t)(r0 + rbase + i) * D_ + col] = acc[j][i];
  }
}

// ---------------------------------------------------------------------------
extern "C" void kernel_launch(void* const* d_in, const int* in_sizes, int n_in,
                              void* d_out, int out_size, void* d_ws, size_t ws_size,
                              hipStream_t stream) {
  (void)in_sizes; (void)n_in; (void)out_size; (void)ws_size;
  const float* x           = (const float*)d_in[0];
  const float* latents     = (const float*)d_in[1];
  const float* mask_x      = (const float*)d_in[2];
  const float* mask_latent = (const float*)d_in[3];
  const float* ln_x_g      = (const float*)d_in[4];
  const float* ln_x_b      = (const float*)d_in[5];
  const float* ln_l_g      = (const float*)d_in[6];
  const float* ln_l_b      = (const float*)d_in[7];
  const float* Wq          = (const float*)d_in[8];
  const float* Wkv         = (const float*)d_in[9];
  const float* qk_gamma    = (const float*)d_in[10];
  const float* Wo          = (const float*)d_in[11];
  float* out = (float*)d_out;

  // workspace carve (f16 buffers, 256B aligned)
  size_t off = 0;
  auto carve = [&](size_t elems) {
    void* r = (char*)d_ws + off;
    off += ((elems * sizeof(_Float16)) + 255) & ~(size_t)255;
    return (_Float16*)r;
  };
  _Float16* xn    = carve((size_t)B_ * SX_ * D_);      // 32 MB
  _Float16* lnh   = carve((size_t)B_ * SL_ * D_);      //  2 MB
  _Float16* Wq_t  = carve((size_t)D_ * D_);            //  2 MB
  _Float16* Wkv_t = carve((size_t)2 * D_ * D_);        //  4 MB
  _Float16* Wo_t  = carve((size_t)D_ * D_);            //  2 MB
  _Float16* qh    = carve((size_t)B_ * SL_ * D_);      //  2 MB
  _Float16* kh    = carve((size_t)B_ * SKV_ * D_);     // 34 MB
  _Float16* vt    = carve((size_t)B_ * SKV_ * D_);     // 34 MB
  _Float16* yh    = carve((size_t)B_ * SL_ * D_);      //  2 MB

  // weights -> f16 transposed
  k_transpose_f16<<<D_,     256, 0, stream>>>(Wq,  Wq_t,  D_, D_);
  k_transpose_f16<<<2 * D_, 256, 0, stream>>>(Wkv, Wkv_t, D_, 2 * D_);
  k_transpose_f16<<<D_,     256, 0, stream>>>(Wo,  Wo_t,  D_, D_);

  // layernorms -> f16
  k_layernorm_f16<<<B_ * SX_, 256, 0, stream>>>(x, ln_x_g, ln_x_b, xn);
  k_layernorm_f16<<<B_ * SL_, 256, 0, stream>>>(latents, ln_l_g, ln_l_b, lnh);

  // kv projection (fused k RMSnorm + v transpose), 16x64 per wave
  k_gemm_kv<<<dim3(B_ * SKV_ / 64, 32), 128, 0, stream>>>(xn, lnh, Wkv_t, qk_gamma, kh, vt);

  // q projection (fused RMSnorm), 16x64 per wave
  k_gemm_q<<<dim3(B_ * SL_ / 64, H_), 128, 0, stream>>>(lnh, Wq_t, qk_gamma, qh);

  // attention
  k_attn<<<B_ * H_ * (SL_ / 16), 128, 0, stream>>>(qh, kh, vt, mask_x, mask_latent, yh);

  // output projection -> f32
  k_gemm_out<<<dim3(B_ * SL_ / 64, D_ / 64), 128, 0, stream>>>(yh, Wo_t, out);
}
